// KokoroModel_59854664237192
// MI455X (gfx1250) — compile-verified, matrix operand order checked by reading, not run on hardware
//
#include <hip/hip_runtime.h>
#include <cstdint>
#include <cstddef>

// ---------------------------------------------------------------------------
// CDNA5 WMMA (wave32): D(16x16,f32) = A(16x32,f16) * B(32x16,f16) + C
// 16-bit operand layout: lane = (half, idx); VGPR0..3 hold k = 8*half..8*half+7,
// VGPR4..7 hold k = 16+8*half..16+8*half+7  -> two contiguous 8xf16 (b128) chunks.
// C/D layout: VGPR r holds row r+8*half, col = idx.
// ---------------------------------------------------------------------------
typedef __attribute__((ext_vector_type(16))) _Float16 v16h;
typedef __attribute__((ext_vector_type(8)))  _Float16 h8;
typedef __attribute__((ext_vector_type(8)))  float    v8f;

#define DI __device__ __forceinline__

DI v16h comb(h8 lo, h8 hi) {
  return __builtin_shufflevector(lo, hi, 0, 1, 2, 3, 4, 5, 6, 7, 8, 9, 10, 11, 12, 13, 14, 15);
}
DI float sigm(float x) { return 1.0f / (1.0f + expf(-x)); }
DI v8f wmma(v16h a, v16h b, v8f c) {
  return __builtin_amdgcn_wmma_f32_16x16x32_f16(false, a, false, b, (short)0, c, false, false);
}

// ---------------------------------------------------------------------------
// GEMM: D[M,N] = act( A[M,K] * W[N,K]^T + bias (+bias2) )
// A f16 (optionally split along K), W f16 padded to 32-multiples (no guards in
// the K loop). One wave computes a 32x32 tile (4 WMMAs / 8 b128 loads per step).
// Optional f32 and/or f16 destinations.
// ---------------------------------------------------------------------------
DI void gemm_kseg(const _Float16* __restrict__ A, int lda,
                  const _Float16* __restrict__ Wp, int ldw, int len,
                  int mg0, int mg1, int ng0, int ng1, int half,
                  v8f& c00, v8f& c01, v8f& c10, v8f& c11)
{
  const _Float16* pa0 = A + (size_t)mg0 * lda + (half << 3);
  const _Float16* pa1 = A + (size_t)mg1 * lda + (half << 3);
  const _Float16* pb0 = Wp + (size_t)ng0 * ldw + (half << 3);
  const _Float16* pb1 = Wp + (size_t)ng1 * ldw + (half << 3);
  for (int kk = 0; kk < len; kk += 32) {
    h8 a0l = *(const h8*)(pa0 + kk), a0h = *(const h8*)(pa0 + kk + 16);
    h8 a1l = *(const h8*)(pa1 + kk), a1h = *(const h8*)(pa1 + kk + 16);
    h8 b0l = *(const h8*)(pb0 + kk), b0h = *(const h8*)(pb0 + kk + 16);
    h8 b1l = *(const h8*)(pb1 + kk), b1h = *(const h8*)(pb1 + kk + 16);
    v16h a0 = comb(a0l, a0h), a1 = comb(a1l, a1h);
    v16h b0 = comb(b0l, b0h), b1 = comb(b1l, b1h);
    c00 = wmma(a0, b0, c00);
    c01 = wmma(a0, b1, c01);
    c10 = wmma(a1, b0, c10);
    c11 = wmma(a1, b1, c11);
  }
}

__global__ __launch_bounds__(256) void gemm_kernel(
    const _Float16* __restrict__ A0, const _Float16* __restrict__ A1, int K0,
    const _Float16* __restrict__ W, int ldw,
    const float* __restrict__ bias, const float* __restrict__ bias2,
    float* __restrict__ D32, _Float16* __restrict__ D16,
    int M, int N, int K, int relu, int Nt, int tiles)
{
  int wave = threadIdx.x >> 5, lane = threadIdx.x & 31;
  int half = lane >> 4, idx = lane & 15;
  int tile = blockIdx.x * 8 + wave;
  if (tile >= tiles) return;
  int mt = tile / Nt, nt = tile - mt * Nt;
  int m0 = mt << 5, n0 = nt << 5;
  int mg0 = m0 + idx, mg1 = m0 + 16 + idx;
  int ng0 = n0 + idx, ng1 = n0 + 16 + idx;
  v8f c00 = {}, c01 = {}, c10 = {}, c11 = {};
  gemm_kseg(A0, K0, W, ldw, K0, mg0, mg1, ng0, ng1, half, c00, c01, c10, c11);
  if (A1) gemm_kseg(A1, K - K0, W + K0, ldw, K - K0, mg0, mg1, ng0, ng1, half, c00, c01, c10, c11);
#pragma unroll
  for (int ni = 0; ni < 2; ++ni) {
    int col = n0 + (ni << 4) + idx;
    if (col >= N) continue;
    float bb = bias ? bias[col] : 0.f;
    if (bias2) bb += bias2[col];
#pragma unroll
    for (int mi = 0; mi < 2; ++mi) {
      v8f acc = (ni == 0) ? (mi == 0 ? c00 : c10) : (mi == 0 ? c01 : c11);
#pragma unroll
      for (int r = 0; r < 8; ++r) {
        int row = m0 + (mi << 4) + r + (half << 3);
        if (row < M) {
          float vv = acc[r] + bb;
          if (relu) vv = fmaxf(vv, 0.f);
          if (D32) D32[(size_t)row * N + col] = vv;
          if (D16) D16[(size_t)row * N + col] = (_Float16)vv;
        }
      }
    }
  }
}

// ---------------------------------------------------------------------------
// Persistent LSTM scan (one workgroup per LSTM, time loop internal).
// xg = x@W_ih^T + b_ih + b_hh precomputed (f32 [8][T][2048]); per step:
//   g = xg[:,t,:] + h @ W_hh^T ; i,f,g,o gates ; c,h update.
// h f16 in LDS (WMMA A source, b128 reads), c in regs, gates via LDS f32.
// ---------------------------------------------------------------------------
struct Lstm {
  const float*    xg;   // [8][T][2048] f32
  const _Float16* whh;  // [2048][512] f16
  _Float16*       hs;   // [8][T][512] f16
  int T;
  int rev;
};

#define HS 520  // padded f16 row stride: 16B-aligned b128 rows, conflict-free banks

__global__ __launch_bounds__(512) void lstm_scan_kernel(Lstm La, Lstm Lb) {
  Lstm L = (blockIdx.x == 0) ? La : Lb;
  extern __shared__ char smem[];
  _Float16* h16 = (_Float16*)smem;                 // [16][HS], rows 8..15 stay 0
  float* gacc   = (float*)(smem + 16 * HS * 2);    // [8][2048]
  const int tid = threadIdx.x;
  const int wave = tid >> 5, lane = tid & 31, half = lane >> 4, idx = lane & 15;
  for (int i = tid; i < 16 * HS; i += 512) h16[i] = (_Float16)0.f;
  float cr[8];
#pragma unroll
  for (int u = 0; u < 8; ++u) cr[u] = 0.f;
  __syncthreads();
  const _Float16* ha = &h16[idx * HS + (half << 3)];
  for (int s = 0; s < L.T; ++s) {
    int t = L.rev ? (L.T - 1 - s) : s;
    // Phase A: 16 waves x 8 tiles cover N = 2048 gate columns, M = batch(8..16)
    for (int j = 0; j < 8; ++j) {
      int n0 = ((wave << 3) + j) << 4;
      const _Float16* wb = L.whh + (size_t)(n0 + idx) * 512 + (half << 3);
      v8f acc = {};
      for (int kk = 0; kk < 512; kk += 32) {
        h8 al = *(const h8*)(ha + kk), ah = *(const h8*)(ha + kk + 16);
        h8 bl = *(const h8*)(wb + kk), bh = *(const h8*)(wb + kk + 16);
        acc = wmma(comb(al, ah), comb(bl, bh), acc);
      }
      if (half == 0) {  // C rows 0..7 = valid batches
        int col = n0 + idx;
#pragma unroll
        for (int r = 0; r < 8; ++r)
          gacc[r * 2048 + col] = acc[r] + L.xg[((size_t)r * L.T + t) * 2048 + col];
      }
    }
    __syncthreads();
    // Phase B: gate nonlinearities + state update (thread tid owns (b=u, j=tid))
#pragma unroll
    for (int u = 0; u < 8; ++u) {
      float gi = gacc[u * 2048 + tid];
      float gf = gacc[u * 2048 + 512 + tid];
      float gg = gacc[u * 2048 + 1024 + tid];
      float go = gacc[u * 2048 + 1536 + tid];
      float c = sigm(gf) * cr[u] + sigm(gi) * tanhf(gg);
      cr[u] = c;
      _Float16 hv = (_Float16)(sigm(go) * tanhf(c));
      h16[u * HS + tid] = hv;
      L.hs[((size_t)u * L.T + t) * 512 + tid] = hv;
    }
    __syncthreads();
  }
}

// ---------------------------------------------------------------------------
// Flash attention, one wave per (b, head, 16-query tile).
// S = QK^T/8 (+pad mask), online softmax, ctx = P V. All operands f16.
// ---------------------------------------------------------------------------
#define PS 40  // Ps f16 row stride: 16B-aligned, conflict-free

__global__ __launch_bounds__(32) void attn_kernel(
    const _Float16* __restrict__ q, const _Float16* __restrict__ kbuf,
    const _Float16* __restrict__ vt, const int* __restrict__ pad,
    _Float16* __restrict__ ctx)
{
  __shared__ _Float16 Ps[16 * PS];   // P tile bounce (C-layout -> A-layout)
  const int lane = threadIdx.x & 31, half = lane >> 4, idx = lane & 15;
  const int qt = blockIdx.x, h = blockIdx.y, b = blockIdx.z;
  const int q0 = qt << 4;
  v16h aq[2];
  {
    const _Float16* pq = q + (size_t)(b * 1024 + q0 + idx) * 512 + h * 64 + (half << 3);
#pragma unroll
    for (int c = 0; c < 2; ++c) {
      h8 lo = *(const h8*)(pq + (c << 5));
      h8 hi = *(const h8*)(pq + (c << 5) + 16);
      aq[c] = comb(lo, hi);
    }
  }
  v8f acc0 = {}, acc1 = {}, acc2 = {}, acc3 = {};
  float mrow[8], lrow[8];
#pragma unroll
  for (int r = 0; r < 8; ++r) { mrow[r] = -3.0e38f; lrow[r] = 0.f; }
  const _Float16* pp = &Ps[idx * PS + (half << 3)];
  const _Float16* pv = vt + (size_t)((b * 8 + h) * 64 + idx) * 1024 + (half << 3);
  for (int kt = 0; kt < 32; ++kt) {           // 32 keys per iteration
    int kbase = kt << 5;
    float s0[8], s1[8];
#pragma unroll
    for (int jt = 0; jt < 2; ++jt) {
      v8f sa = {};
      int key = b * 1024 + kbase + (jt << 4) + idx;
      const _Float16* pk = kbuf + (size_t)key * 512 + h * 64 + (half << 3);
#pragma unroll
      for (int c = 0; c < 2; ++c) {
        h8 lo = *(const h8*)(pk + (c << 5));
        h8 hi = *(const h8*)(pk + (c << 5) + 16);
        sa = wmma(aq[c], comb(lo, hi), sa);
      }
      int pm = pad[key];
      if (jt == 0) {
#pragma unroll
        for (int r = 0; r < 8; ++r) s0[r] = pm ? -1.0e9f : sa[r] * 0.125f;
      } else {
#pragma unroll
        for (int r = 0; r < 8; ++r) s1[r] = pm ? -1.0e9f : sa[r] * 0.125f;
      }
    }
    // Online softmax per row (rows r+8*half live in fixed 16-lane groups)
#pragma unroll
    for (int r = 0; r < 8; ++r) {
      float mx = fmaxf(s0[r], s1[r]);
      mx = fmaxf(mx, __shfl_xor(mx, 1, 16));
      mx = fmaxf(mx, __shfl_xor(mx, 2, 16));
      mx = fmaxf(mx, __shfl_xor(mx, 4, 16));
      mx = fmaxf(mx, __shfl_xor(mx, 8, 16));
      float mnew = fmaxf(mrow[r], mx);
      float sf = expf(mrow[r] - mnew);
      mrow[r] = mnew;
      acc0[r] *= sf; acc1[r] *= sf; acc2[r] *= sf; acc3[r] *= sf;
      float e0 = expf(s0[r] - mnew), e1 = expf(s1[r] - mnew);
      float rs = e0 + e1;
      rs += __shfl_xor(rs, 1, 16);
      rs += __shfl_xor(rs, 2, 16);
      rs += __shfl_xor(rs, 4, 16);
      rs += __shfl_xor(rs, 8, 16);
      lrow[r] = lrow[r] * sf + rs;
      int row = r + (half << 3);
      Ps[row * PS + idx]      = (_Float16)e0;
      Ps[row * PS + 16 + idx] = (_Float16)e1;
    }
    asm volatile("s_wait_dscnt 0" ::: "memory");  // intra-wave LDS RAW
    v16h pa;
    {
      h8 lo = *(const h8*)(pp);
      h8 hi = *(const h8*)(pp + 16);
      pa = comb(lo, hi);
    }
#pragma unroll
    for (int j = 0; j < 4; ++j) {
      h8 lo = *(const h8*)(pv + (size_t)(j << 4) * 1024 + kbase);
      h8 hi = *(const h8*)(pv + (size_t)(j << 4) * 1024 + kbase + 16);
      v8f& a = (j == 0) ? acc0 : (j == 1) ? acc1 : (j == 2) ? acc2 : acc3;
      a = wmma(pa, comb(lo, hi), a);
    }
  }
#pragma unroll
  for (int r = 0; r < 8; ++r) {
    int row = r + (half << 3);
    float inv = 1.0f / lrow[r];
    size_t base = (size_t)(b * 1024 + q0 + row) * 512 + h * 64;
    ctx[base + idx]      = (_Float16)(acc0[r] * inv);
    ctx[base + 16 + idx] = (_Float16)(acc1[r] * inv);
    ctx[base + 32 + idx] = (_Float16)(acc2[r] * inv);
    ctx[base + 48 + idx] = (_Float16)(acc3[r] * inv);
  }
}

// ---------------------------------------------------------------------------
// Elementwise glue kernels
// ---------------------------------------------------------------------------
__global__ void cvt_pad_kernel(const float* __restrict__ s, _Float16* __restrict__ d,
                               int N, int K, int Np, int Kp) {
  int i = blockIdx.x * 256 + threadIdx.x;
  if (i >= Np * Kp) return;
  int n = i / Kp, k = i - n * Kp;
  d[i] = (n < N && k < K) ? (_Float16)s[(size_t)n * K + k] : (_Float16)0.f;
}

__global__ void embed_kernel(const int* __restrict__ idx, const float* __restrict__ emb,
                             _Float16* __restrict__ out, int n) {
  int i = blockIdx.x * 256 + threadIdx.x;
  if (i < n) out[i] = (_Float16)emb[(size_t)idx[i >> 9] * 512 + (i & 511)];
}

__global__ void melshift_kernel(const float* __restrict__ mel, _Float16* __restrict__ out) {
  int i = blockIdx.x * 256 + threadIdx.x;
  if (i >= 8 * 1024 * 96) return;
  int c = i % 96, bt = i / 96, t = bt & 1023, b = bt >> 10;
  float v = 0.f;
  if (c < 80 && t > 0) v = mel[(size_t)(b * 1024 + t - 1) * 80 + c];
  out[i] = (_Float16)v;
}

__global__ void cum_kernel(const float* __restrict__ dur, int* __restrict__ cum) {
  int b = threadIdx.x;
  if (b >= 8) return;
  int acc = 0;
  for (int j = 0; j < 256; ++j) { acc += (int)rintf(dur[b * 256 + j]); cum[b * 256 + j] = acc; }
}

__global__ void lr_index_kernel(const int* __restrict__ cum, int* __restrict__ idxmap,
                                int* __restrict__ pad) {
  int i = blockIdx.x * 256 + threadIdx.x;
  if (i >= 8 * 1024) return;
  int b = i >> 10, p = i & 1023;
  const int* c = cum + b * 256;
  int lo = 0, hi = 256;
  while (lo < hi) { int mid = (lo + hi) >> 1; if (c[mid] <= p) lo = mid + 1; else hi = mid; }
  pad[i] = (p >= c[255]) ? 1 : 0;
  idxmap[i] = lo < 256 ? lo : 255;
}

__global__ void expand_kernel(const _Float16* __restrict__ enc, const int* __restrict__ idxmap,
                              const int* __restrict__ pad, _Float16* __restrict__ out) {
  int i = blockIdx.x * 256 + threadIdx.x;
  if (i >= 8 * 1024 * 512) return;
  int c = i & 511, row = i >> 9;          // row = b*1024 + p
  int b = row >> 10;
  out[i] = pad[row] ? (_Float16)0.f : enc[(size_t)(b * 256 + idxmap[row]) * 512 + c];
}

__global__ void vtrans_kernel(const _Float16* __restrict__ v, _Float16* __restrict__ vt) {
  int i = blockIdx.x * 256 + threadIdx.x;
  if (i >= 8 * 8 * 64 * 1024) return;
  int p = i & 1023, f = (i >> 10) & 63, h = (i >> 16) & 7, b = i >> 19;
  vt[i] = v[(size_t)(b * 1024 + p) * 512 + h * 64 + f];
}

// ---------------------------------------------------------------------------
// Host side
// ---------------------------------------------------------------------------
static void launch_gemm(hipStream_t st, const _Float16* A0, const _Float16* A1, int K0,
                        const _Float16* W, int ldw, const float* bias, const float* bias2,
                        float* D32, _Float16* D16, int M, int N, int K, int relu) {
  int Mt = (M + 31) >> 5, Nt = (N + 31) >> 5;
  int tiles = Mt * Nt;
  int blocks = (tiles + 7) >> 3;
  gemm_kernel<<<blocks, 256, 0, st>>>(A0, A1, K0, W, ldw, bias, bias2, D32, D16,
                                      M, N, K, relu, Nt, tiles);
}

extern "C" void kernel_launch(void* const* d_in, const int* in_sizes, int n_in,
                              void* d_out, int out_size, void* d_ws, size_t ws_size,
                              hipStream_t stream) {
  (void)in_sizes; (void)n_in; (void)out_size; (void)ws_size;
  const int*   ph_idx = (const int*)d_in[0];
  const float* mel    = (const float*)d_in[1];
  const float* durs   = (const float*)d_in[2];
  const float* emb    = (const float*)d_in[3];
  const float* b_ih_f = (const float*)d_in[6];
  const float* b_hh_f = (const float*)d_in[7];
  const float* b_ih_b = (const float*)d_in[10];
  const float* b_hh_b = (const float*)d_in[11];
  const float* proj_b = (const float*)d_in[13];
  const float* dur_b1 = (const float*)d_in[15];
  const float* dur_b2 = (const float*)d_in[17];
  const float* dur_b3 = (const float*)d_in[19];
  const float* melin_b = (const float*)d_in[21];
  const float* attin_b = (const float*)d_in[23];
  const float* attout_b = (const float*)d_in[25];
  const float* dec_b_ih = (const float*)d_in[28];
  const float* dec_b_hh = (const float*)d_in[29];
  const float* melout_b = (const float*)d_in[31];
  float* out = (float*)d_out;

  char* ws = (char*)d_ws;
  size_t off = 0;
  auto alloc = [&](size_t bytes) -> char* {
    char* p = ws + off; off = (off + bytes + 255) & ~(size_t)255; return p;
  };
  // f32 [N][K] weight -> f16 [Np][Kp] zero-padded (32-multiples) for guard-free K loops
  auto cvt = [&](int in_idx, int N, int K, int Np, int Kp) -> _Float16* {
    _Float16* dst = (_Float16*)alloc((size_t)Np * Kp * sizeof(_Float16));
    int n = Np * Kp;
    cvt_pad_kernel<<<(n + 255) / 256, 256, 0, stream>>>((const float*)d_in[in_idx], dst, N, K, Np, Kp);
    return dst;
  };

  // --- weights to padded f16 ---
  _Float16* wihf16   = cvt(4, 2048, 512, 2048, 512);
  _Float16* whhf16   = cvt(5, 2048, 512, 2048, 512);
  _Float16* wihb16   = cvt(8, 2048, 512, 2048, 512);
  _Float16* whhb16   = cvt(9, 2048, 512, 2048, 512);
  _Float16* projw16  = cvt(12, 512, 1024, 512, 1024);
  _Float16* durw116  = cvt(14, 512, 512, 512, 512);
  _Float16* durw216  = cvt(16, 256, 512, 256, 512);
  _Float16* durw316  = cvt(18, 1, 256, 32, 256);       // N padded to 32
  _Float16* melinw16 = cvt(20, 512, 80, 512, 96);      // K padded to 96
  _Float16* attinw16 = cvt(22, 1536, 512, 1536, 512);
  _Float16* attoutw16 = cvt(24, 512, 512, 512, 512);
  _Float16* decwih16 = cvt(26, 2048, 1024, 2048, 1024);
  _Float16* decwhh16 = cvt(27, 2048, 512, 2048, 512);
  _Float16* meloutw16 = cvt(30, 80, 512, 96, 512);     // N padded to 96

  // --- activations: f16 everywhere except xg gate pre-activations + outputs ---
  _Float16* xemb16  = (_Float16*)alloc((size_t)2048 * 512 * 2);
  float*    xgf     = (float*)alloc((size_t)2048 * 2048 * 4);
  float*    xgb     = (float*)alloc((size_t)2048 * 2048 * 4);
  _Float16* hf16    = (_Float16*)alloc((size_t)2048 * 512 * 2);
  _Float16* hb16    = (_Float16*)alloc((size_t)2048 * 512 * 2);
  _Float16* enc16   = (_Float16*)alloc((size_t)2048 * 512 * 2);
  _Float16* h1b16   = (_Float16*)alloc((size_t)2048 * 512 * 2);
  _Float16* h2b16   = (_Float16*)alloc((size_t)2048 * 256 * 2);
  int*      cumb    = (int*)alloc((size_t)8 * 256 * 4);
  int*      idxm    = (int*)alloc((size_t)8192 * 4);
  int*      padm    = (int*)alloc((size_t)8192 * 4);
  _Float16* exp16   = (_Float16*)alloc((size_t)8192 * 512 * 2);
  _Float16* mshift16 = (_Float16*)alloc((size_t)8192 * 96 * 2);
  _Float16* qin16   = (_Float16*)alloc((size_t)8192 * 512 * 2);
  _Float16* qb16    = (_Float16*)alloc((size_t)8192 * 512 * 2);
  _Float16* kb16    = (_Float16*)alloc((size_t)8192 * 512 * 2);
  _Float16* vb16    = (_Float16*)alloc((size_t)8192 * 512 * 2);
  _Float16* vt16    = (_Float16*)alloc((size_t)4194304 * 2);
  _Float16* ctx16   = (_Float16*)alloc((size_t)8192 * 512 * 2);
  _Float16* att16   = (_Float16*)alloc((size_t)8192 * 512 * 2);
  float*    xgd     = (float*)alloc((size_t)8192 * 2048 * 4);
  _Float16* decout16 = (_Float16*)alloc((size_t)8192 * 512 * 2);

  const int SMEM = 16 * HS * 2 + 8 * 2048 * 4;  // h16 + gacc

  // 1) embedding (f16)
  embed_kernel<<<4096, 256, 0, stream>>>(ph_idx, emb, xemb16, 2048 * 512);
  // 2) encoder input gates (b_ih + b_hh folded into bias)
  launch_gemm(stream, xemb16, nullptr, 512, wihf16, 512, b_ih_f, b_hh_f, xgf, nullptr, 2048, 2048, 512, 0);
  launch_gemm(stream, xemb16, nullptr, 512, wihb16, 512, b_ih_b, b_hh_b, xgb, nullptr, 2048, 2048, 512, 0);
  // 3) encoder scans (fwd + bwd as two concurrent workgroups)
  {
    Lstm Lf{xgf, whhf16, hf16, 256, 0};
    Lstm Lb{xgb, whhb16, hb16, 256, 1};
    lstm_scan_kernel<<<2, 512, SMEM, stream>>>(Lf, Lb);
  }
  // 4) projection over concat(hf,hb)
  launch_gemm(stream, hf16, hb16, 512, projw16, 1024, proj_b, nullptr, nullptr, enc16, 2048, 512, 1024, 0);
  // 5) duration predictor
  launch_gemm(stream, enc16, nullptr, 512, durw116, 512, dur_b1, nullptr, nullptr, h1b16, 2048, 512, 512, 1);
  launch_gemm(stream, h1b16, nullptr, 512, durw216, 512, dur_b2, nullptr, nullptr, h2b16, 2048, 256, 512, 1);
  launch_gemm(stream, h2b16, nullptr, 256, durw316, 256, dur_b3, nullptr, out + 8 * 1024 * 80, nullptr, 2048, 1, 256, 0);
  // 6) length regulation
  cum_kernel<<<1, 8, 0, stream>>>(durs, cumb);
  lr_index_kernel<<<32, 256, 0, stream>>>(cumb, idxm, padm);
  expand_kernel<<<16384, 256, 0, stream>>>(enc16, idxm, padm, exp16);
  // 7) teacher-forced decoder inputs (mel shifted, K padded 80->96)
  melshift_kernel<<<3072, 256, 0, stream>>>(mel, mshift16);
  launch_gemm(stream, mshift16, nullptr, 96, melinw16, 96, melin_b, nullptr, nullptr, qin16, 8192, 512, 96, 0);
  // 8) Q/K/V projections (f16 outputs feed attention directly)
  launch_gemm(stream, qin16, nullptr, 512, attinw16, 512, attin_b, nullptr, nullptr, qb16, 8192, 512, 512, 0);
  launch_gemm(stream, exp16, nullptr, 512, attinw16 + (size_t)512 * 512, 512, attin_b + 512, nullptr, nullptr, kb16, 8192, 512, 512, 0);
  launch_gemm(stream, exp16, nullptr, 512, attinw16 + (size_t)1024 * 512, 512, attin_b + 1024, nullptr, nullptr, vb16, 8192, 512, 512, 0);
  vtrans_kernel<<<16384, 256, 0, stream>>>(vb16, vt16);
  // 9) attention
  attn_kernel<<<dim3(64, 8, 8), 32, 0, stream>>>(qb16, kb16, vt16, padm, ctx16);
  launch_gemm(stream, ctx16, nullptr, 512, attoutw16, 512, attout_b, nullptr, nullptr, att16, 8192, 512, 512, 0);
  // 10) decoder LSTM over concat(qin, attended)
  launch_gemm(stream, qin16, att16, 512, decwih16, 1024, dec_b_ih, dec_b_hh, xgd, nullptr, 8192, 2048, 1024, 0);
  {
    Lstm Ld{xgd, decwhh16, decout16, 1024, 0};
    lstm_scan_kernel<<<1, 512, SMEM, stream>>>(Ld, Ld);
  }
  // 11) mel output (f32, straight into d_out)
  launch_gemm(stream, decout16, nullptr, 512, meloutw16, 512, melout_b, nullptr, out, nullptr, 8192, 80, 512, 0);
}
